// LocationSensitiveAttention_67808943669734
// MI455X (gfx1250) — matrix operand order repaired and checked by reading, not compiled, not run on hardware
//
#include <hip/hip_runtime.h>
#include <hip/hip_bf16.h>
#include <math.h>

// Problem constants (from reference)
constexpr int Bn = 128;   // batch
constexpr int Tn = 512;   // time
constexpr int En = 512;   // encoder dim
constexpr int Dn = 1024;  // decoder dim
constexpr int An = 128;   // attention dim
constexpr int Fn = 32;    // location filters
constexpr int KC = 31;    // conv kernel

typedef __attribute__((ext_vector_type(16))) __bf16 v16bf;
typedef __attribute__((ext_vector_type(8)))  float  v8f;

struct bfrag32 { uint4 lo, hi; };   // 32 bytes == v16bf

__device__ __forceinline__ __bf16 f2bf(float f) {
    union { float f; unsigned u; } v; v.f = f;
    unsigned r = v.u + 0x7FFFu + ((v.u >> 16) & 1u);   // round-to-nearest-even
    unsigned short h = (unsigned short)(r >> 16);
    return __builtin_bit_cast(__bf16, h);
}

// ---------------------------------------------------------------------------
// Prep 1: dec_p[b,a] = sum_d dec[b,d] * W_dec[a,d]      (tiny, L2-resident)
// ---------------------------------------------------------------------------
__global__ __launch_bounds__(128) void dec_proj_kernel(
    const float* __restrict__ dec, const float* __restrict__ Wdec,
    float* __restrict__ dec_p)
{
    int b = blockIdx.x, a = threadIdx.x;
    const float4* dv = (const float4*)(dec  + (size_t)b * Dn);
    const float4* wv = (const float4*)(Wdec + (size_t)a * Dn);
    float acc = 0.f;
#pragma unroll 4
    for (int i = 0; i < Dn / 4; ++i) {
        float4 x = dv[i], w = wv[i];
        acc += x.x * w.x + x.y * w.y + x.z * w.z + x.w * w.w;
    }
    dec_p[b * An + a] = acc;
}

// ---------------------------------------------------------------------------
// Prep 2: pre-swizzle W_enc (fp32 [A,E]) into bf16 WMMA B-fragments.
// B matrix is 32(K) x 16(N) per tile; per-lane layout (ISA 7.12.2 mirror of
// the documented 8/16-bit B layouts): lane l holds column n=l&15,
// element j -> k=(l>>4)*8+j (j<8), k=16+(l>>4)*8+(j-8) (j>=8).
// Fragment storage: WencF[((kb*8+nt)*32+lane)*16 + j]  (32 B per lane, coalesced)
// ---------------------------------------------------------------------------
__global__ __launch_bounds__(256) void wenc_frag_kernel(
    const float* __restrict__ Wenc, __bf16* __restrict__ WencF)
{
    int kb = blockIdx.x;             // 0..15 : K-block of 32 over E
    int nt = threadIdx.x >> 5;       // 0..7  : N tile of 16 over A
    int l  = threadIdx.x & 31;
    int n  = l & 15;
    int a  = nt * 16 + n;
    __bf16* dst = WencF + ((size_t)(kb * 8 + nt) * 32 + l) * 16;
#pragma unroll
    for (int j = 0; j < 16; ++j) {
        int k = (l >> 4) * 8 + j + (j >= 8 ? 8 : 0);
        int e = kb * 32 + k;
        dst[j] = f2bf(Wenc[(size_t)a * En + e]);
    }
}

// ---------------------------------------------------------------------------
// Prep 3: Weff[a,k] = sum_f W_loc[a,f]*conv_w[f,0,k], emitted directly as
// bf16 B-fragments (K padded 31 -> 32 with zeros).
// ---------------------------------------------------------------------------
__global__ __launch_bounds__(256) void weff_frag_kernel(
    const float* __restrict__ Wloc, const float* __restrict__ convw,
    __bf16* __restrict__ WeffF)
{
    int nt = threadIdx.x >> 5;       // 0..7
    int l  = threadIdx.x & 31;
    int n  = l & 15;
    int a  = nt * 16 + n;
    __bf16* dst = WeffF + ((size_t)(nt * 32 + l)) * 16;
#pragma unroll
    for (int j = 0; j < 16; ++j) {
        int k = (l >> 4) * 8 + j + (j >= 8 ? 8 : 0);
        float acc = 0.f;
        if (k < KC) {
#pragma unroll
            for (int f = 0; f < Fn; ++f)
                acc += Wloc[a * Fn + f] * convw[f * KC + k];
        }
        dst[j] = f2bf(acc);
    }
}

// ---------------------------------------------------------------------------
// Main: energies[b, t0..t0+15].  8 waves/WG, wave w owns A-columns w*16..+15.
//   C[16t x 16a] += enc_tile(bf16) x WencF   (16 WMMAs over E=512)
//   C            += prev_window(bf16) x WeffF (1 WMMA, location term)
//   e[t] = b_e + sum_a We[a]*tanh(C[t,a] + dec_p[b,a])
// ---------------------------------------------------------------------------
__global__ __launch_bounds__(256) void energies_kernel(
    const float* __restrict__ enc, const float* __restrict__ prev,
    const __bf16* __restrict__ WencF, const __bf16* __restrict__ WeffF,
    const float* __restrict__ dec_p, const float* __restrict__ We,
    const float* __restrict__ be, float* __restrict__ energies)
{
    constexpr int PITCH = En + 8;            // +8 bf16 -> 4-bank lane stride
    __shared__ __bf16 sA[16 * PITCH];        // 16 x 512 bf16 tile (~16.6 KB)
    __shared__ float  sPrev[64];             // prev window [t0-15, t0+48]
    __shared__ float  sRed[16];              // per-row energy accumulators

    const int b   = blockIdx.y;
    const int t0  = blockIdx.x * 16;
    const int tid = threadIdx.x;
    const int wave = tid >> 5, lane = tid & 31;
    const int m  = lane & 15;                // row within tile (A-frag / C)
    const int kh = lane >> 4;                // K-half selector

    // ---- stage encoder tile (fp32 -> bf16), fully coalesced float4 loads
    const float4* encB = (const float4*)(enc + ((size_t)b * Tn + t0) * En);
#pragma unroll
    for (int i = 0; i < 8; ++i) {
        int idx = i * 256 + tid;             // 0..2047 float4s
        int row = idx >> 7, c4 = idx & 127;
        float4 v = encB[row * (En / 4) + c4];
        __bf16* p = &sA[row * PITCH + c4 * 4];
        p[0] = f2bf(v.x); p[1] = f2bf(v.y); p[2] = f2bf(v.z); p[3] = f2bf(v.w);
    }
    if (tid < 64) {
        int t = t0 - 15 + tid;
        sPrev[tid] = (t >= 0 && t < Tn) ? prev[(size_t)b * Tn + t] : 0.f;
    }
    if (tid < 16) sRed[tid] = 0.f;
    __syncthreads();

    // ---- 16x16 f32 accumulator tile
    v8f c = {};

    // A-fragment (16x32 bf16, ISA 7.12.2): lane<16 -> K {0..7,16..23},
    // lane>=16 -> K {8..15,24..31}; two aligned 16 B LDS chunks per kb.
#pragma unroll 8
    for (int kb = 0; kb < En / 32; ++kb) {
        const __bf16* p0 = &sA[m * PITCH + kb * 32 + kh * 8];
        bfrag32 fr;
        fr.lo = *(const uint4*)(p0);
        fr.hi = *(const uint4*)(p0 + 16);
        v16bf aF = __builtin_bit_cast(v16bf, fr);
        v16bf bF = *(const v16bf*)(WencF + ((size_t)(kb * 8 + wave) * 32 + lane) * 16);
        c = __builtin_amdgcn_wmma_f32_16x16x32_bf16(false, aF, false, bF,
                                                    (short)0, c, false, false);
    }

    // ---- location term: X[m,k] = prev[t0+m-15+k] = sPrev[m+k], k<31
    {
        v16bf xF;
#pragma unroll
        for (int j = 0; j < 16; ++j) {
            int k = kh * 8 + j + (j >= 8 ? 8 : 0);
            xF[j] = f2bf(k < KC ? sPrev[m + k] : 0.f);
        }
        v16bf wF = *(const v16bf*)(WeffF + ((size_t)wave * 32 + lane) * 16);
        c = __builtin_amdgcn_wmma_f32_16x16x32_bf16(false, xF, false, wF,
                                                    (short)0, c, false, false);
    }

    // ---- epilogue: +dec_p, tanh, *We, reduce over a
    const int a  = wave * 16 + (lane & 15);
    const float dv = dec_p[b * An + a];
    const float we = We[a];
#pragma unroll
    for (int r = 0; r < 8; ++r) {
        // C layout: lane<16 -> (M=r, N=lane); lane>=16 -> (M=r+8, N=lane-16)
        float v = tanhf(c[r] + dv) * we;
        v += __shfl_xor(v, 1, 32);
        v += __shfl_xor(v, 2, 32);
        v += __shfl_xor(v, 4, 32);
        v += __shfl_xor(v, 8, 32);           // sum over the 16 lanes of a half
        if ((lane & 15) == 0)
            atomicAdd(&sRed[r + kh * 8], v); // ds_add_f32 across 8 waves
    }
    __syncthreads();
    if (tid < 16)
        energies[(size_t)b * Tn + t0 + tid] = sRed[tid] + be[0];
}

// ---------------------------------------------------------------------------
// Softmax over T per batch row
// ---------------------------------------------------------------------------
__global__ __launch_bounds__(256) void softmax_kernel(
    const float* __restrict__ energies, float* __restrict__ weights)
{
    __shared__ float sred[256];
    int b = blockIdx.x, tid = threadIdx.x;
    const float* e = energies + (size_t)b * Tn;
    float v0 = e[tid], v1 = e[tid + 256];
    sred[tid] = fmaxf(v0, v1);
    __syncthreads();
    for (int s = 128; s > 0; s >>= 1) {
        if (tid < s) sred[tid] = fmaxf(sred[tid], sred[tid + s]);
        __syncthreads();
    }
    float mx = sred[0];
    __syncthreads();
    float e0 = __expf(v0 - mx), e1 = __expf(v1 - mx);
    sred[tid] = e0 + e1;
    __syncthreads();
    for (int s = 128; s > 0; s >>= 1) {
        if (tid < s) sred[tid] += sred[tid + s];
        __syncthreads();
    }
    float inv = 1.f / sred[0];
    weights[(size_t)b * Tn + tid]       = e0 * inv;
    weights[(size_t)b * Tn + tid + 256] = e1 * inv;
}

// ---------------------------------------------------------------------------
// context[b,e] = sum_t w[b,t] * enc[b,t,e]   (memory-bound, coalesced)
// ---------------------------------------------------------------------------
__global__ __launch_bounds__(256) void context_kernel(
    const float* __restrict__ enc, const float* __restrict__ weights,
    float* __restrict__ ctx)
{
    __shared__ float sw[Tn];
    int b = blockIdx.x, tid = threadIdx.x;
    sw[tid]       = weights[(size_t)b * Tn + tid];
    sw[tid + 256] = weights[(size_t)b * Tn + tid + 256];
    __syncthreads();
    const float* eb = enc + (size_t)b * Tn * En;
    float a0 = 0.f, a1 = 0.f;
    for (int t = 0; t < Tn; ++t) {
        float w = sw[t];
        a0 += w * eb[(size_t)t * En + tid];
        a1 += w * eb[(size_t)t * En + tid + 256];
    }
    ctx[(size_t)b * En + tid]       = a0;
    ctx[(size_t)b * En + tid + 256] = a1;
}

// ---------------------------------------------------------------------------
extern "C" void kernel_launch(void* const* d_in, const int* in_sizes, int n_in,
                              void* d_out, int out_size, void* d_ws, size_t ws_size,
                              hipStream_t stream)
{
    (void)in_sizes; (void)n_in; (void)out_size; (void)ws_size;
    const float* enc   = (const float*)d_in[0];  // [B,T,E]
    const float* dec   = (const float*)d_in[1];  // [B,D]
    const float* prev  = (const float*)d_in[2];  // [B,T]
    const float* Wenc  = (const float*)d_in[3];  // [A,E]
    const float* Wdec  = (const float*)d_in[4];  // [A,D]
    const float* convw = (const float*)d_in[5];  // [F,1,K]
    const float* Wloc  = (const float*)d_in[6];  // [A,F]
    const float* We    = (const float*)d_in[7];  // [1,A]
    const float* be    = (const float*)d_in[8];  // [1]

    float* ctx     = (float*)d_out;              // [B,E] first
    float* weights = (float*)d_out + Bn * En;    // [B,T] second

    char* ws = (char*)d_ws;
    float*  dec_p    = (float*)(ws);                   //  64 KB
    float*  energies = (float*)(ws + (64 << 10));      // 256 KB
    __bf16* WencF    = (__bf16*)(ws + (320 << 10));    // 128 KB
    __bf16* WeffF    = (__bf16*)(ws + (448 << 10));    //   8 KB

    dec_proj_kernel <<<dim3(Bn), dim3(An),  0, stream>>>(dec, Wdec, dec_p);
    wenc_frag_kernel<<<dim3(En / 32), dim3(256), 0, stream>>>(Wenc, WencF);
    weff_frag_kernel<<<dim3(1),  dim3(256), 0, stream>>>(Wloc, convw, WeffF);
    energies_kernel <<<dim3(Tn / 16, Bn), dim3(256), 0, stream>>>(
        enc, prev, WencF, WeffF, dec_p, We, be, energies);
    softmax_kernel  <<<dim3(Bn), dim3(256), 0, stream>>>(energies, weights);
    context_kernel  <<<dim3(Bn), dim3(256), 0, stream>>>(enc, weights, ctx);
}
